// RotationNet_549755814609
// MI455X (gfx1250) — matrix operand order repaired and checked by reading, not compiled, not run on hardware
//
#include <hip/hip_runtime.h>

// ---------------------------------------------------------------------------
// Problem constants (reference: B=512, C=2048, H=W=8, J=30, CF=512)
// ---------------------------------------------------------------------------
#define BSZ   512
#define CIN   2048
#define HW    64          // 8x8
#define CF    512
#define NJ    30
#define DRAW  15450       // J*(CF+3)
#define DPAD  15456       // padded to multiple of 32 (K for final WMMA GEMM)
#define NPAD  160         // 6 root + 126 pose + 28 zero pad (10 N-tiles of 16)

// d_out layout (flat, in return order)
#define OFF_ROOT  0                 // 512*6
#define OFF_POSE  3072              // 512*126
#define OFF_SHAPE 67584             // 512*10
#define OFF_CAM   72704             // 512*3

// workspace layout (bytes, all 16B aligned)
#define WS_FBUF   0                         // bf16 f, (b,hw,o): 512*64*512*2 = 33554432
#define WS_FEAT   33554432                  // bf16 feat, (b, DPAD): 512*15456*2 = 15826944
#define WS_WPACK  49381376                  // bf16 wpack, (NPAD, DPAD): 160*15456*2 = 4945920

typedef __attribute__((ext_vector_type(16))) __bf16 v16bf;
typedef __attribute__((ext_vector_type(8)))  float  v8f;

union Frag { v16bf v; uint4 q[2]; };

__device__ __forceinline__ unsigned short f2bf(float x) {
  unsigned u = __float_as_uint(x);
  unsigned r = u + 0x7FFFu + ((u >> 16) & 1u);   // round-to-nearest-even
  return (unsigned short)(r >> 16);
}
__device__ __forceinline__ unsigned pk2(float a, float b) {
  return (unsigned)f2bf(a) | ((unsigned)f2bf(b) << 16);
}
__device__ __forceinline__ float bf2f(unsigned short h) {
  return __uint_as_float(((unsigned)h) << 16);
}

// ---------------------------------------------------------------------------
// Kernel 1: 1x1 conv (WMMA GEMM) + BN(eval) + ReLU.
// One block per batch: O_b(512x64) = conv_w(512x2048) . X_b(2048x64).
// 512 threads = 16 waves; wave w owns M-tiles {2w, 2w+1} x all 4 N-tiles.
// f stored bf16 in (b, hw, o) layout for the coalesced gather in kernel 4.
// ---------------------------------------------------------------------------
__global__ __launch_bounds__(512) void conv_bn_relu_gemm(
    const float* __restrict__ img_feat, const float* __restrict__ conv_w,
    const float* __restrict__ conv_b,  const float* __restrict__ gamma,
    const float* __restrict__ beta,    const float* __restrict__ mean,
    const float* __restrict__ var,     unsigned short* __restrict__ fbuf) {
  __shared__ __align__(16) unsigned short Wlds[CF * 72];   // 512 rows x 64 K, pitch 72
  __shared__ __align__(16) unsigned short Xlds[HW * 72];   // 64 (hw) rows x 64 K, pitch 72

  const int tid  = threadIdx.x;
  const int lane = tid & 31;
  const int wv   = tid >> 5;               // 0..15
  const int b    = blockIdx.x;
  const float* xbase = img_feat + (size_t)b * CIN * HW;

  v8f acc[2][4];
#pragma unroll
  for (int i = 0; i < 2; ++i)
#pragma unroll
    for (int t = 0; t < 4; ++t)
#pragma unroll
      for (int r = 0; r < 8; ++r) acc[i][t][r] = 0.f;

  for (int k0 = 0; k0 < CIN; k0 += 64) {
    // ---- stage W chunk: 512(o) x 64(k), fp32 -> bf16, row-major along K ----
#pragma unroll
    for (int i = 0; i < 8; ++i) {
      int g  = tid + i * 512;              // 4096 groups of 8 elements
      int o  = g >> 3;
      int kk = (g & 7) * 8;
      const float4* p = (const float4*)(conv_w + (size_t)o * CIN + k0 + kk);
      float4 l0 = p[0], l1 = p[1];
      uint4 pk;
      pk.x = pk2(l0.x, l0.y); pk.y = pk2(l0.z, l0.w);
      pk.z = pk2(l1.x, l1.y); pk.w = pk2(l1.z, l1.w);
      *(uint4*)&Wlds[o * 72 + kk] = pk;
    }
    // ---- stage X chunk transposed: Xlds[n][k] from X_b[k][n] ----
#pragma unroll
    for (int i = 0; i < 2; ++i) {
      int g  = tid + i * 512;              // 1024 groups of 4 (along n)
      int kk = g >> 4;
      int n4 = (g & 15) * 4;
      float4 l = *(const float4*)(xbase + (size_t)(k0 + kk) * HW + n4);
      Xlds[(n4 + 0) * 72 + kk] = f2bf(l.x);
      Xlds[(n4 + 1) * 72 + kk] = f2bf(l.y);
      Xlds[(n4 + 2) * 72 + kk] = f2bf(l.z);
      Xlds[(n4 + 3) * 72 + kk] = f2bf(l.w);
    }
    __syncthreads();

#pragma unroll
    for (int kk0 = 0; kk0 < 64; kk0 += 32) {
      Frag a[2], bm[4];
      const int ka = kk0 + ((lane < 16) ? 0 : 8);    // A: 16x32 bf16 lane layout
      const int kb = kk0 + ((lane < 16) ? 0 : 8);    // B: 32x16 bf16 lane layout
#pragma unroll
      for (int i = 0; i < 2; ++i) {
        int o = (wv * 2 + i) * 16 + (lane & 15);
        a[i].q[0] = *(const uint4*)&Wlds[o * 72 + ka];
        a[i].q[1] = *(const uint4*)&Wlds[o * 72 + ka + 16];
      }
#pragma unroll
      for (int t = 0; t < 4; ++t) {
        int n = t * 16 + (lane & 15);
        bm[t].q[0] = *(const uint4*)&Xlds[n * 72 + kb];
        bm[t].q[1] = *(const uint4*)&Xlds[n * 72 + kb + 16];
      }
#pragma unroll
      for (int i = 0; i < 2; ++i)
#pragma unroll
        for (int t = 0; t < 4; ++t)
          acc[i][t] = __builtin_amdgcn_wmma_f32_16x16x32_bf16(
              false, a[i].v, false, bm[t].v, (short)0, acc[i][t], false, false);
    }
    __syncthreads();
  }

  // ---- epilogue: bias + BN + ReLU, pack 8 consecutive o per lane, b128 store ----
#pragma unroll
  for (int i = 0; i < 2; ++i) {
    const int ob = wv * 32 + i * 16 + ((lane < 16) ? 0 : 8);  // first of 8 o's
    float s[8], c[8];
#pragma unroll
    for (int r = 0; r < 8; ++r) {
      int o = ob + r;
      float sc = gamma[o] * rsqrtf(var[o] + 1e-5f);
      s[r] = sc;
      c[r] = (conv_b[o] - mean[o]) * sc + beta[o];
    }
#pragma unroll
    for (int t = 0; t < 4; ++t) {
      int hw = t * 16 + (lane & 15);
      float v[8];
#pragma unroll
      for (int r = 0; r < 8; ++r) v[r] = fmaxf(acc[i][t][r] * s[r] + c[r], 0.f);
      uint4 pk;
      pk.x = pk2(v[0], v[1]); pk.y = pk2(v[2], v[3]);
      pk.z = pk2(v[4], v[5]); pk.w = pk2(v[6], v[7]);
      *(uint4*)&fbuf[((size_t)b * HW + hw) * CF + ob] = pk;
    }
  }
}

// ---------------------------------------------------------------------------
// Kernel 2: global average pool + shape (10) & cam (3) FC heads. Block per b.
// ---------------------------------------------------------------------------
__global__ __launch_bounds__(256) void pool_heads(
    const float* __restrict__ img_feat,
    const float* __restrict__ shape_w, const float* __restrict__ shape_b,
    const float* __restrict__ cam_w,   const float* __restrict__ cam_b,
    float* __restrict__ out) {
  __shared__ float pooled[CIN];
  __shared__ float red[256];
  const int b = blockIdx.x, tid = threadIdx.x;
  const float* base = img_feat + (size_t)b * CIN * HW;
#pragma unroll
  for (int i = 0; i < 8; ++i) {
    int c = tid + i * 256;
    const float4* p = (const float4*)(base + (size_t)c * HW);
    float sum = 0.f;
#pragma unroll
    for (int q = 0; q < 16; ++q) { float4 v = p[q]; sum += v.x + v.y + v.z + v.w; }
    pooled[c] = sum * (1.f / 64.f);
  }
  __syncthreads();
  for (int o = 0; o < 13; ++o) {
    const float* wrow = (o < 10) ? (shape_w + (size_t)o * CIN)
                                 : (cam_w + (size_t)(o - 10) * CIN);
    float p = 0.f;
#pragma unroll
    for (int i = 0; i < 8; ++i) { int c = tid + i * 256; p += pooled[c] * wrow[c]; }
    red[tid] = p;
    __syncthreads();
    for (int s = 128; s > 0; s >>= 1) {
      if (tid < s) red[tid] += red[tid + s];
      __syncthreads();
    }
    if (tid == 0) {
      if (o < 10) out[OFF_SHAPE + b * 10 + o] = red[0] + shape_b[o];
      else        out[OFF_CAM + b * 3 + (o - 10)] = red[0] + cam_b[o - 10];
    }
    __syncthreads();
  }
}

// ---------------------------------------------------------------------------
// Kernel 3: pack [root_w ; pose_w ; zero] into bf16 (NPAD x DPAD), K-major.
// ---------------------------------------------------------------------------
__global__ __launch_bounds__(256) void pack_weights(
    const float* __restrict__ root_w, const float* __restrict__ pose_w,
    unsigned short* __restrict__ wpack) {
  const int n = blockIdx.x;                       // 0..159
  for (int k = threadIdx.x; k < DPAD; k += 256) {
    float v = 0.f;
    if (k < DRAW) {
      if (n < 6)        v = root_w[(size_t)n * DRAW + k];
      else if (n < 132) v = pose_w[(size_t)(n - 6) * DRAW + k];
    }
    wpack[(size_t)n * DPAD + k] = f2bf(v);
  }
}

// ---------------------------------------------------------------------------
// Kernel 4: bilinear joint gather + concat coords -> bf16 feat (512 x DPAD).
// Block per batch; channel reads are coalesced ((b,hw,o) layout of f).
// ---------------------------------------------------------------------------
__global__ __launch_bounds__(256) void gather_feat(
    const unsigned short* __restrict__ fbuf, const float* __restrict__ coords,
    unsigned short* __restrict__ featpad) {
  const int b = blockIdx.x, tid = threadIdx.x;
  const size_t fb = (size_t)b * HW * CF;
  const size_t ob = (size_t)b * DPAD;
  for (int j = 0; j < NJ; ++j) {
    float x = coords[b * (NJ * 3) + j * 3 + 0];
    float y = coords[b * (NJ * 3) + j * 3 + 1];
    float x0f = floorf(x), y0f = floorf(y);
    float wx1 = x - x0f, wy1 = y - y0f;
    float wx0 = 1.f - wx1, wy0 = 1.f - wy1;
    int x0 = (int)x0f, y0 = (int)y0f, x1 = x0 + 1, y1 = y0 + 1;
    int   hwc[4];
    float wt[4];
    int   yis[4] = {y0, y0, y1, y1};
    int   xis[4] = {x0, x1, x0, x1};
    float wws[4] = {wy0 * wx0, wy0 * wx1, wy1 * wx0, wy1 * wx1};
#pragma unroll
    for (int c = 0; c < 4; ++c) {
      int xi = xis[c], yi = yis[c];
      bool valid = (xi >= 0) && (xi < 8) && (yi >= 0) && (yi < 8);
      int xc = xi < 0 ? 0 : (xi > 7 ? 7 : xi);
      int yc = yi < 0 ? 0 : (yi > 7 ? 7 : yi);
      hwc[c] = yc * 8 + xc;
      wt[c]  = valid ? wws[c] : 0.f;
    }
#pragma unroll
    for (int it = 0; it < 2; ++it) {
      int cf = tid + it * 256;
      float v = 0.f;
#pragma unroll
      for (int c = 0; c < 4; ++c)
        v += wt[c] * bf2f(fbuf[fb + (size_t)hwc[c] * CF + cf]);
      featpad[ob + j * (CF + 3) + cf] = f2bf(v);
    }
  }
  if (tid < NJ * 3) {
    int j = tid / 3, c = tid % 3;
    featpad[ob + j * (CF + 3) + CF + c] = f2bf(coords[b * (NJ * 3) + tid]);
  }
  if (tid >= 96 && tid < 96 + (DPAD - DRAW))
    featpad[ob + DRAW + (tid - 96)] = 0;
}

// ---------------------------------------------------------------------------
// Kernel 5: final WMMA GEMM: out(512 x 160) = feat(512 x DPAD) . wpack^T.
// A/B fragments loaded straight from global (row-major along K == native
// per-lane WMMA layout). Bias fused; root/pose scattered into d_out.
// 8 blocks (M-tile 64) x 256 threads (8 waves): wave = (mt, 5 N-tiles).
// ---------------------------------------------------------------------------
__global__ __launch_bounds__(256) void final_gemm(
    const unsigned short* __restrict__ feat, const unsigned short* __restrict__ wpack,
    const float* __restrict__ root_b, const float* __restrict__ pose_b,
    float* __restrict__ out) {
  const int tid  = threadIdx.x;
  const int lane = tid & 31;
  const int wv   = tid >> 5;
  const int mt   = wv & 3;                 // M-tile within block
  const int ng   = wv >> 2;                // N-tile group (0 or 1), 5 tiles each
  const int mbase = blockIdx.x * 64 + mt * 16;

  v8f acc[5];
#pragma unroll
  for (int t = 0; t < 5; ++t)
#pragma unroll
    for (int r = 0; r < 8; ++r) acc[t][r] = 0.f;

  const int m = mbase + (lane & 15);
  const uint4* arow = (const uint4*)(feat + (size_t)m * DPAD);
  const uint4* brow[5];
#pragma unroll
  for (int t = 0; t < 5; ++t) {
    int n = (ng * 5 + t) * 16 + (lane & 15);
    brow[t] = (const uint4*)(wpack + (size_t)n * DPAD);
  }

  for (int k0 = 0; k0 < DPAD; k0 += 32) {
    Frag a;
    const int ka = k0 + ((lane < 16) ? 0 : 8);
    a.q[0] = arow[ka >> 3];
    a.q[1] = arow[(ka >> 3) + 2];
    const int kb = k0 + ((lane < 16) ? 0 : 8);
#pragma unroll
    for (int t = 0; t < 5; ++t) {
      Frag bm;
      bm.q[0] = brow[t][kb >> 3];
      bm.q[1] = brow[t][(kb >> 3) + 2];
      acc[t] = __builtin_amdgcn_wmma_f32_16x16x32_bf16(
          false, a.v, false, bm.v, (short)0, acc[t], false, false);
    }
  }

#pragma unroll
  for (int t = 0; t < 5; ++t) {
    int n = (ng * 5 + t) * 16 + (lane & 15);
    if (n >= 132) continue;
    float bias = (n < 6) ? root_b[n] : pose_b[n - 6];
#pragma unroll
    for (int r = 0; r < 8; ++r) {
      int bb = mbase + r + ((lane < 16) ? 0 : 8);
      float v = acc[t][r] + bias;
      if (n < 6) out[OFF_ROOT + bb * 6 + n] = v;
      else       out[OFF_POSE + bb * 126 + (n - 6)] = v;
    }
  }
}

// ---------------------------------------------------------------------------
extern "C" void kernel_launch(void* const* d_in, const int* in_sizes, int n_in,
                              void* d_out, int out_size, void* d_ws, size_t ws_size,
                              hipStream_t stream) {
  (void)in_sizes; (void)n_in; (void)out_size; (void)ws_size;
  const float* img_feat = (const float*)d_in[0];
  const float* joints   = (const float*)d_in[1];
  const float* conv_w   = (const float*)d_in[2];
  const float* conv_b   = (const float*)d_in[3];
  const float* gamma    = (const float*)d_in[4];
  const float* beta     = (const float*)d_in[5];
  const float* mean     = (const float*)d_in[6];
  const float* var      = (const float*)d_in[7];
  const float* root_w   = (const float*)d_in[8];
  const float* root_b   = (const float*)d_in[9];
  const float* pose_w   = (const float*)d_in[10];
  const float* pose_b   = (const float*)d_in[11];
  const float* shape_w  = (const float*)d_in[12];
  const float* shape_b  = (const float*)d_in[13];
  const float* cam_w    = (const float*)d_in[14];
  const float* cam_b    = (const float*)d_in[15];
  float* out = (float*)d_out;

  unsigned short* fbuf    = (unsigned short*)((char*)d_ws + WS_FBUF);
  unsigned short* featpad = (unsigned short*)((char*)d_ws + WS_FEAT);
  unsigned short* wpack   = (unsigned short*)((char*)d_ws + WS_WPACK);

  conv_bn_relu_gemm<<<BSZ, 512, 0, stream>>>(img_feat, conv_w, conv_b, gamma,
                                             beta, mean, var, fbuf);
  pool_heads<<<BSZ, 256, 0, stream>>>(img_feat, shape_w, shape_b, cam_w, cam_b, out);
  pack_weights<<<NPAD, 256, 0, stream>>>(root_w, pose_w, wpack);
  gather_feat<<<BSZ, 256, 0, stream>>>(fbuf, joints, featpad);
  final_gemm<<<BSZ / 64, 256, 0, stream>>>(featpad, wpack, root_b, pose_b, out);
}
